// SequentialSSM_15607911153852
// MI455X (gfx1250) — compile-verified, matrix operand order checked by reading, not run on hardware
//
#include <hip/hip_runtime.h>
#include <hip/hip_bf16.h>

// ---------------------------------------------------------------------------
// SequentialSSM for MI455X (gfx1250, wave32, WMMA)
//   B=8, T=4096, H=1024, S=16, M = B*T = 32768
//
// Pipeline:
//   0. convert x -> bf16; transpose-convert W_in/W_D/W_out -> bf16 [N][K]
//   1. fold W_BB = W_B @ B_mat (1024x16), bBB = b_B @ B_mat
//   2. pack gate weights [W_BB | W_ig | W_fg | 0] -> WgT bf16 [128][1024]
//   3. GEMM1 (WMMA): xp = x @ W_in + b_in          -> xp_f32 + xp_bf16
//   4. GEMM_G (WMMA): gates = xp @ Wg + bg          -> Bx, sigmoid->ig, fg
//   5. chunked parallel scan: h_t = fg*h + ig*Bx    -> hs
//   6. hsC = hs @ C_mat (K=16, VALU)
//   7. GEMM_D (WMMA): y1 = tanh(xp @ W_D + b_D + hsC) -> y1_bf16
//   8. GEMM_O (WMMA): y = y1 @ W_out + b_out + xp     -> y_f32
//   9. LayerNorm rows of 1024 -> d_out
//
// GEMM uses double-buffered LDS staging via GLOBAL_LOAD_ASYNC_TO_LDS_B128
// (ASYNCcnt-tracked, no VGPR round-trip) with a sync fallback.
// ---------------------------------------------------------------------------

typedef unsigned short ushort_t;
typedef __attribute__((ext_vector_type(16))) __bf16 v16bf;
typedef __attribute__((ext_vector_type(8)))  float  v8f;
typedef __attribute__((ext_vector_type(4)))  int    v4i_;

#define CB 8
#define CT 4096
#define CH_ 1024
#define CS 16
#define CM (CB * CT)          // 32768
#define SCAN_CHUNKS 64
#define SCAN_LEN    64        // CT / SCAN_CHUNKS

#define TILE_M 128
#define TILE_N 128
#define TILE_K 32
#define LDSP   TILE_K         // lds row pitch in bf16 elements

#if defined(__has_builtin)
#if __has_builtin(__builtin_amdgcn_global_load_async_to_lds_b128) && \
    __has_builtin(__builtin_amdgcn_s_wait_asynccnt)
#define USE_ASYNC_LDS 1
#endif
#endif
#ifndef USE_ASYNC_LDS
#define USE_ASYNC_LDS 0
#endif

#define AS1 __attribute__((address_space(1)))
#define AS3 __attribute__((address_space(3)))

struct bf16x16 { uint4 a, b; };   // 32 bytes == v16bf

__device__ __forceinline__ ushort_t f2bf(float f) {
  unsigned u = __builtin_bit_cast(unsigned, f);
  unsigned r = u + 0x7FFFu + ((u >> 16) & 1u);
  return (ushort_t)(r >> 16);
}

// ---------------------------------------------------------------------------
// small prep kernels
// ---------------------------------------------------------------------------
__global__ void k_f32_to_bf16(const float* __restrict__ in,
                              ushort_t* __restrict__ out, int n) {
  int i = blockIdx.x * blockDim.x + threadIdx.x;
  if (i < n) out[i] = f2bf(in[i]);
}

// W: [K][N] row-major f32  ->  WT: [N][K] bf16
__global__ void k_convT(const float* __restrict__ W, ushort_t* __restrict__ WT,
                        int K, int N) {
  int idx = blockIdx.x * blockDim.x + threadIdx.x;
  if (idx >= K * N) return;
  int k = idx / N, n = idx % N;
  WT[(size_t)n * K + k] = f2bf(W[idx]);
}

// W_BB[h][s] = sum_j W_B[h][j] * B_mat[j][s]; bBB[s] = sum_j b_B[j]*B_mat[j][s]
__global__ void k_fold_WB(const float* __restrict__ W_B,
                          const float* __restrict__ b_B,
                          const float* __restrict__ B_mat,
                          float* __restrict__ W_BB, float* __restrict__ bBB) {
  int idx = blockIdx.x * blockDim.x + threadIdx.x;  // H*S
  if (idx >= CH_ * CS) return;
  int h = idx >> 4, s = idx & 15;
  float acc = 0.f;
  for (int j = 0; j < CH_; ++j)
    acc += W_B[(size_t)h * CH_ + j] * B_mat[j * CS + s];
  W_BB[idx] = acc;
  if (h == 0) {
    float a = 0.f;
    for (int j = 0; j < CH_; ++j) a += b_B[j] * B_mat[j * CS + s];
    bBB[s] = a;
  }
}

// WgT bf16 [128][1024]: rows 0..15 = W_BB^T, 16..31 = W_ig^T, 32..47 = W_fg^T
__global__ void k_pack_gates(const float* __restrict__ W_BB,
                             const float* __restrict__ bBB,
                             const float* __restrict__ W_ig,
                             const float* __restrict__ b_ig,
                             const float* __restrict__ W_fg,
                             const float* __restrict__ b_fg,
                             ushort_t* __restrict__ WgT,
                             float* __restrict__ biasg) {
  int idx = blockIdx.x * blockDim.x + threadIdx.x;  // 128*1024
  if (idx >= 128 * CH_) return;
  int n = idx / CH_, k = idx % CH_;
  float v = 0.f;
  if (n < 16)       v = W_BB[k * CS + n];
  else if (n < 32)  v = W_ig[k * CS + (n - 16)];
  else if (n < 48)  v = W_fg[k * CS + (n - 32)];
  WgT[(size_t)n * CH_ + k] = f2bf(v);
  if (k == 0) {
    float bv = 0.f;
    if (n < 16)      bv = bBB[n];
    else if (n < 32) bv = b_ig[n - 16];
    else if (n < 48) bv = b_fg[n - 32];
    biasg[n] = bv;
  }
}

// ---------------------------------------------------------------------------
// WMMA GEMM: D[M,N] = A[M,K](bf16) x B^T-stored[N,K](bf16) + epilogue
//   EPI 0: out_f32 = acc+bias; out_bf16 = bf16(same)               (xp)
//   EPI 1: gates: n<16 Bx, n<32 sigmoid->aux0(ig), n<48 ->aux1(fg)
//   EPI 2: out_bf16 = bf16(tanh(acc + bias + extra))               (y1)
//   EPI 3: out_f32 = acc + bias + extra                            (y pre-LN)
// block = 256 threads (8 waves), block tile 128x128, k-step 32,
// double-buffered LDS (2 x 8KB per operand).
// wave (wid&3) -> 32 rows, (wid>>2) -> 64 cols: 2x4 wmma accumulators
// ---------------------------------------------------------------------------
template <int EPI>
__global__ __launch_bounds__(256) void k_gemm(
    const ushort_t* __restrict__ A, const ushort_t* __restrict__ BT,
    const float* __restrict__ bias, const float* __restrict__ extra,
    float* __restrict__ out_f32, ushort_t* __restrict__ out_bf16,
    float* __restrict__ aux0, float* __restrict__ aux1,
    int M, int N, int K) {
  __shared__ ushort_t sA[2][TILE_M * LDSP];
  __shared__ ushort_t sB[2][TILE_N * LDSP];

  const int tid  = threadIdx.x;
  const int bm   = blockIdx.y * TILE_M;
  const int bn   = blockIdx.x * TILE_N;
  const int wid  = tid >> 5;
  const int lane = tid & 31;
  const int lo   = lane & 15;
  const int hi   = lane >> 4;
  const int wrow = (wid & 3) * 32;   // 0,32,64,96
  const int wcol = (wid >> 2) * 64;  // 0,64

  // stage one 128x32 bf16 tile of A and of BT into LDS buffer `buf`
  auto stage = [&](int kt, int buf) {
#pragma unroll
    for (int i = 0; i < 2; ++i) {
      int idx = tid + i * 256;        // 0..511
      int r = idx >> 2, seg = idx & 3;
      const ushort_t* ga = A + (size_t)(bm + r) * K + kt + seg * 8;
      const ushort_t* gb = BT + (size_t)(bn + r) * K + kt + seg * 8;
      ushort_t* la = &sA[buf][r * LDSP + seg * 8];
      ushort_t* lb = &sB[buf][r * LDSP + seg * 8];
#if USE_ASYNC_LDS
      __builtin_amdgcn_global_load_async_to_lds_b128((AS1 v4i_*)ga,
                                                     (AS3 v4i_*)la, 0, 0);
      __builtin_amdgcn_global_load_async_to_lds_b128((AS1 v4i_*)gb,
                                                     (AS3 v4i_*)lb, 0, 0);
#else
      *(uint4*)la = *(const uint4*)ga;
      *(uint4*)lb = *(const uint4*)gb;
#endif
    }
  };

  v8f acc[2][4];
#pragma unroll
  for (int mi = 0; mi < 2; ++mi)
#pragma unroll
    for (int ni = 0; ni < 4; ++ni) acc[mi][ni] = (v8f)0.f;

  stage(0, 0);  // prologue

  for (int kt = 0; kt < K; kt += TILE_K) {
    const int cur = (kt >> 5) & 1;
#if USE_ASYNC_LDS
    __builtin_amdgcn_s_wait_asynccnt(0);  // this wave's tile portion landed
#endif
    __syncthreads();                      // whole tile visible to all waves

    if (kt + TILE_K < K) {
      // safe: end-barrier of iteration kt-1 ordered all reads of buf cur^1
      stage(kt + TILE_K, cur ^ 1);
#if !USE_ASYNC_LDS
      __builtin_prefetch(A + (size_t)(bm + (tid >> 1)) * K + kt + TILE_K, 0, 1);
      __builtin_prefetch(BT + (size_t)(bn + (tid >> 1)) * K + kt + TILE_K, 0, 1);
#endif
    }

    // fragments: lane lo = row-in-tile, hi selects K 8-block halves
    v16bf af[2], bfr[4];
#pragma unroll
    for (int mi = 0; mi < 2; ++mi) {
      const ushort_t* p = &sA[cur][(wrow + mi * 16 + lo) * LDSP + hi * 8];
      bf16x16 t;
      t.a = *(const uint4*)p;         // K = hi*8 + 0..7
      t.b = *(const uint4*)(p + 16);  // K = 16 + hi*8 + 0..7
      af[mi] = __builtin_bit_cast(v16bf, t);
    }
#pragma unroll
    for (int ni = 0; ni < 4; ++ni) {
      const ushort_t* p = &sB[cur][(wcol + ni * 16 + lo) * LDSP + hi * 8];
      bf16x16 t;
      t.a = *(const uint4*)p;
      t.b = *(const uint4*)(p + 16);
      bfr[ni] = __builtin_bit_cast(v16bf, t);
    }

#pragma unroll
    for (int mi = 0; mi < 2; ++mi)
#pragma unroll
      for (int ni = 0; ni < 4; ++ni)
        acc[mi][ni] = __builtin_amdgcn_wmma_f32_16x16x32_bf16(
            false, af[mi], false, bfr[ni], (short)0, acc[mi][ni], false, false);

    __syncthreads();  // all waves done reading buf `cur` before it is reused
  }

  // epilogue: acc element v, lane (lo,hi) -> m = base + hi*8 + v, n = base + lo
#pragma unroll
  for (int mi = 0; mi < 2; ++mi) {
#pragma unroll
    for (int ni = 0; ni < 4; ++ni) {
#pragma unroll
      for (int v = 0; v < 8; ++v) {
        int m = bm + wrow + mi * 16 + hi * 8 + v;
        int n = bn + wcol + ni * 16 + lo;
        float val = acc[mi][ni][v] + bias[n];
        if constexpr (EPI == 0) {
          out_f32[(size_t)m * N + n]  = val;
          out_bf16[(size_t)m * N + n] = f2bf(val);
        } else if constexpr (EPI == 1) {
          if (n < 16)
            out_f32[(size_t)m * CS + n] = val;
          else if (n < 32)
            aux0[(size_t)m * CS + (n - 16)] = 1.f / (1.f + __expf(-val));
          else if (n < 48)
            aux1[(size_t)m * CS + (n - 32)] = 1.f / (1.f + __expf(-val));
        } else if constexpr (EPI == 2) {
          float t = tanhf(val + extra[(size_t)m * N + n]);
          out_bf16[(size_t)m * N + n] = f2bf(t);
        } else {  // EPI == 3
          out_f32[(size_t)m * N + n] = val + extra[(size_t)m * N + n];
        }
      }
    }
  }
}

// ---------------------------------------------------------------------------
// chunked parallel scan over T: h_t = fg_t*h + ig_t*Bx_t  (per b,s chain)
// ---------------------------------------------------------------------------
__global__ void k_scan_chunk(const float* __restrict__ fg,
                             const float* __restrict__ ig,
                             const float* __restrict__ Bx,
                             float* __restrict__ cA, float* __restrict__ cB) {
  int idx = blockIdx.x * blockDim.x + threadIdx.x;  // b*1024 + c*16 + s
  if (idx >= CB * SCAN_CHUNKS * CS) return;
  int s = idx & 15, c = (idx >> 4) & (SCAN_CHUNKS - 1), b = idx >> 10;
  float A = 1.f, Bc = 0.f;
  size_t base = ((size_t)b * CT + c * SCAN_LEN) * CS + s;
  for (int i = 0; i < SCAN_LEN; ++i) {
    float f = fg[base + i * CS];
    float g = ig[base + i * CS];
    float x = Bx[base + i * CS];
    A *= f;
    Bc = f * Bc + g * x;
  }
  cA[idx] = A;
  cB[idx] = Bc;
}

__global__ void k_scan_carry(const float* __restrict__ cA,
                             const float* __restrict__ cB,
                             float* __restrict__ hin) {
  int idx = blockIdx.x * blockDim.x + threadIdx.x;  // b*16 + s
  if (idx >= CB * CS) return;
  int s = idx & 15, b = idx >> 4;
  float h = 0.f;
  for (int c = 0; c < SCAN_CHUNKS; ++c) {
    int j = (b * SCAN_CHUNKS + c) * CS + s;
    hin[j] = h;
    h = cA[j] * h + cB[j];
  }
}

__global__ void k_scan_apply(const float* __restrict__ fg,
                             const float* __restrict__ ig,
                             const float* __restrict__ Bx,
                             const float* __restrict__ hin,
                             float* __restrict__ hs) {
  int idx = blockIdx.x * blockDim.x + threadIdx.x;
  if (idx >= CB * SCAN_CHUNKS * CS) return;
  int s = idx & 15, c = (idx >> 4) & (SCAN_CHUNKS - 1), b = idx >> 10;
  float h = hin[idx];
  size_t base = ((size_t)b * CT + c * SCAN_LEN) * CS + s;
  for (int i = 0; i < SCAN_LEN; ++i) {
    float f = fg[base + i * CS];
    float g = ig[base + i * CS];
    float x = Bx[base + i * CS];
    h = f * h + g * x;
    hs[base + i * CS] = h;
  }
}

// hsC[m][n] = sum_s hs[m][s] * C_mat[s][n]   (K=16)
__global__ __launch_bounds__(256) void k_hsC(const float* __restrict__ hs,
                                             const float* __restrict__ C,
                                             float* __restrict__ out) {
  int m = blockIdx.x;
  __shared__ float sh[CS];
  if (threadIdx.x < CS) sh[threadIdx.x] = hs[(size_t)m * CS + threadIdx.x];
  __syncthreads();
#pragma unroll
  for (int j = 0; j < CH_ / 256; ++j) {
    int n = threadIdx.x + j * 256;
    float acc = 0.f;
#pragma unroll
    for (int s = 0; s < CS; ++s) acc += sh[s] * C[s * CH_ + n];
    out[(size_t)m * CH_ + n] = acc;
  }
}

// LayerNorm over rows of 1024
__global__ __launch_bounds__(256) void k_layernorm(
    const float* __restrict__ y, const float* __restrict__ gamma,
    const float* __restrict__ beta, float* __restrict__ out) {
  int m = blockIdx.x;
  int tid = threadIdx.x;
  __shared__ float rs1[256], rs2[256];
  float v[4];
  float s1 = 0.f, s2 = 0.f;
#pragma unroll
  for (int j = 0; j < 4; ++j) {
    v[j] = y[(size_t)m * CH_ + tid + j * 256];
    s1 += v[j];
    s2 += v[j] * v[j];
  }
  rs1[tid] = s1;
  rs2[tid] = s2;
  __syncthreads();
  for (int off = 128; off > 0; off >>= 1) {
    if (tid < off) {
      rs1[tid] += rs1[tid + off];
      rs2[tid] += rs2[tid + off];
    }
    __syncthreads();
  }
  float mu   = rs1[0] * (1.f / CH_);
  float var  = rs2[0] * (1.f / CH_) - mu * mu;
  float rstd = rsqrtf(var + 1e-5f);
#pragma unroll
  for (int j = 0; j < 4; ++j) {
    int n = tid + j * 256;
    out[(size_t)m * CH_ + n] = (v[j] - mu) * rstd * gamma[n] + beta[n];
  }
}

// ---------------------------------------------------------------------------
extern "C" void kernel_launch(void* const* d_in, const int* in_sizes, int n_in,
                              void* d_out, int out_size, void* d_ws,
                              size_t ws_size, hipStream_t stream) {
  const float* x     = (const float*)d_in[0];
  const float* W_in  = (const float*)d_in[1];
  const float* b_in  = (const float*)d_in[2];
  const float* B_mat = (const float*)d_in[3];
  const float* C_mat = (const float*)d_in[4];
  const float* W_B   = (const float*)d_in[5];
  const float* b_B   = (const float*)d_in[6];
  const float* W_D   = (const float*)d_in[7];
  const float* b_D   = (const float*)d_in[8];
  const float* W_out = (const float*)d_in[9];
  const float* b_out = (const float*)d_in[10];
  const float* W_ig  = (const float*)d_in[11];
  const float* b_ig  = (const float*)d_in[12];
  const float* W_fg  = (const float*)d_in[13];
  const float* b_fg  = (const float*)d_in[14];
  const float* gamma = (const float*)d_in[15];
  const float* beta  = (const float*)d_in[16];

  char* ws = (char*)d_ws;
  size_t off = 0;
  auto take = [&](size_t bytes) -> void* {
    void* p = ws + off;
    off += (bytes + 255) & ~(size_t)255;
    return p;
  };

  float*    xp_f32 = (float*)take((size_t)CM * CH_ * 4);
  ushort_t* xp_b   = (ushort_t*)take((size_t)CM * CH_ * 2);
  ushort_t* xb     = (ushort_t*)take((size_t)CM * CH_ * 2);  // x bf16, later y1 bf16
  float*    hsC    = (float*)take((size_t)CM * CH_ * 4);     // later y pre-LN
  ushort_t* WinT   = (ushort_t*)take((size_t)CH_ * CH_ * 2);
  ushort_t* WDT    = (ushort_t*)take((size_t)CH_ * CH_ * 2);
  ushort_t* WoutT  = (ushort_t*)take((size_t)CH_ * CH_ * 2);
  ushort_t* WgT    = (ushort_t*)take((size_t)128 * CH_ * 2);
  float*    biasg  = (float*)take(128 * 4);
  float*    W_BB   = (float*)take((size_t)CH_ * CS * 4);
  float*    bBB    = (float*)take(CS * 4);
  float*    Bx     = (float*)take((size_t)CM * CS * 4);
  float*    ig     = (float*)take((size_t)CM * CS * 4);
  float*    fg     = (float*)take((size_t)CM * CS * 4);
  float*    hs     = (float*)take((size_t)CM * CS * 4);
  float*    cA     = (float*)take((size_t)CB * SCAN_CHUNKS * CS * 4);
  float*    cBuf   = (float*)take((size_t)CB * SCAN_CHUNKS * CS * 4);
  float*    hin    = (float*)take((size_t)CB * SCAN_CHUNKS * CS * 4);
  (void)n_in; (void)in_sizes; (void)out_size; (void)ws_size;

  const int nX = CM * CH_;  // 33554432

  // 0. conversions
  k_f32_to_bf16<<<nX / 256, 256, 0, stream>>>(x, xb, nX);
  k_convT<<<(CH_ * CH_) / 256, 256, 0, stream>>>(W_in, WinT, CH_, CH_);
  k_convT<<<(CH_ * CH_) / 256, 256, 0, stream>>>(W_D, WDT, CH_, CH_);
  k_convT<<<(CH_ * CH_) / 256, 256, 0, stream>>>(W_out, WoutT, CH_, CH_);

  // 1-2. fold W_B@B_mat, pack gate weights
  k_fold_WB<<<(CH_ * CS) / 256, 256, 0, stream>>>(W_B, b_B, B_mat, W_BB, bBB);
  k_pack_gates<<<(128 * CH_) / 256, 256, 0, stream>>>(W_BB, bBB, W_ig, b_ig,
                                                      W_fg, b_fg, WgT, biasg);

  // 3. xp = x @ W_in + b_in
  k_gemm<0><<<dim3(CH_ / TILE_N, CM / TILE_M), 256, 0, stream>>>(
      xb, WinT, b_in, nullptr, xp_f32, xp_b, nullptr, nullptr, CM, CH_, CH_);

  // 4. gates = xp @ [W_BB|W_ig|W_fg] -> Bx, ig, fg
  k_gemm<1><<<dim3(1, CM / TILE_M), 256, 0, stream>>>(
      xp_b, WgT, biasg, nullptr, Bx, nullptr, ig, fg, CM, 128, CH_);

  // 5. scan
  k_scan_chunk<<<(CB * SCAN_CHUNKS * CS) / 256, 256, 0, stream>>>(fg, ig, Bx,
                                                                  cA, cBuf);
  k_scan_carry<<<1, 128, 0, stream>>>(cA, cBuf, hin);
  k_scan_apply<<<(CB * SCAN_CHUNKS * CS) / 256, 256, 0, stream>>>(fg, ig, Bx,
                                                                  hin, hs);

  // 6. hsC = hs @ C_mat
  k_hsC<<<CM, 256, 0, stream>>>(hs, C_mat, hsC);

  // 7. y1 = tanh(xp @ W_D + b_D + hsC)  (bf16 into xb, x no longer needed)
  k_gemm<2><<<dim3(CH_ / TILE_N, CM / TILE_M), 256, 0, stream>>>(
      xp_b, WDT, b_D, hsC, nullptr, xb, nullptr, nullptr, CM, CH_, CH_);

  // 8. y = y1 @ W_out + b_out + xp  (f32 into hsC buffer, hsC consumed)
  k_gemm<3><<<dim3(CH_ / TILE_N, CM / TILE_M), 256, 0, stream>>>(
      xb, WoutT, b_out, xp_f32, hsC, nullptr, nullptr, nullptr, CM, CH_, CH_);

  // 9. LayerNorm -> d_out
  k_layernorm<<<CM, 256, 0, stream>>>(hsC, gamma, beta, (float*)d_out);
}
